// HospitalPANDNNModel_48241072668840
// MI455X (gfx1250) — matrix operand order repaired and checked by reading, not compiled, not run on hardware
//
#include <hip/hip_runtime.h>
#include <hip/hip_bf16.h>

// ---------------------------------------------------------------------------
// HospitalPANDNN forward on gfx1250 (MI455X), f16 WMMA w/ f32 accumulation.
// B=32768, D=H=512, C=128, L=3.
// Roofline: ~138 GFLOP total, ~300MB traffic -> f16 WMMA (K=32/instr) puts
// compute ~tens of us, balanced vs 23.3 TB/s HBM. All GEMM tiling is
// compile-time so the WMMA k-loop is branch-free (no accumulator phi-copies).
// ---------------------------------------------------------------------------

typedef __attribute__((ext_vector_type(16))) _Float16 v16h;
typedef __attribute__((ext_vector_type(8)))  float    v8f;

#define ACT_NONE 0
#define ACT_RELU 1
#define ACT_SIG  2

// ---------------------------------------------------------------------------
// WMMA GEMM:  out = act( affine( A[M,K](f16) @ W[K,N] + bias ) )
// W pre-transposed as Wt[N,K] f16.  Grid: x = M/(64*MT), y = N/(16*NT).
// 128 threads = 4 waves; wave owns MT 16-row strips x NT 16-col tiles.
// Requirements: M % (64*MT) == 0, N % (16*NT) == 0 per grid, K % 32 == 0.
// ---------------------------------------------------------------------------
template<int NT, int MT, int ACT, bool AFFINE, bool WF, bool WH>
__global__ __launch_bounds__(128)
void gemm_wmma(const _Float16* __restrict__ A, int lda,
               const _Float16* __restrict__ Wt,
               const float* __restrict__ bias,
               const float* __restrict__ scale, const float* __restrict__ shift,
               float* __restrict__ outF, int ldo,
               _Float16* __restrict__ outH, int ldoh,
               int K)
{
    const int lane  = threadIdx.x & 31;
    const int wave  = threadIdx.x >> 5;
    const int m0    = (blockIdx.x * 4 + wave) * (16 * MT);
    const int nBase = blockIdx.y * (16 * NT);

    const int col  = lane & 15;   // tile column (B/C/D) or tile row (A)
    const int half = lane >> 4;   // K-interleave selector
    const int aoff = half * 8;    // A frag: halves {0..7,16..23} / {8..15,24..31}
    const int boff = half * 8;    // B frag: halves {0..7,16..23} / {8..15,24..31}

    const _Float16* aRow[MT];
#pragma unroll
    for (int i = 0; i < MT; ++i)
        aRow[i] = A + (size_t)(m0 + i * 16 + col) * lda + aoff;
    const _Float16* bRow[NT];
#pragma unroll
    for (int t = 0; t < NT; ++t)
        bRow[t] = Wt + (size_t)(nBase + t * 16 + col) * K + boff;

    v8f acc[MT][NT] = {};

    for (int kk = 0; kk < K; kk += 32) {
        v16h a[MT];
#pragma unroll
        for (int i = 0; i < MT; ++i) {
            ((float4*)&a[i])[0] = *(const float4*)(aRow[i] + kk);
            ((float4*)&a[i])[1] = *(const float4*)(aRow[i] + kk + 16);
        }
        v16h b[NT];
#pragma unroll
        for (int t = 0; t < NT; ++t) {
            ((float4*)&b[t])[0] = *(const float4*)(bRow[t] + kk);
            ((float4*)&b[t])[1] = *(const float4*)(bRow[t] + kk + 16);
        }
#pragma unroll
        for (int i = 0; i < MT; ++i)
#pragma unroll
            for (int t = 0; t < NT; ++t)
                acc[i][t] = __builtin_amdgcn_wmma_f32_16x16x32_f16(
                    false, a[i], false, b[t], (short)0, acc[i][t], false, false);
    }

    // C/D layout: lane<16 -> col=lane rows +0..7 ; lane>=16 -> rows +8..15.
#pragma unroll
    for (int i = 0; i < MT; ++i) {
        const int mBase = m0 + i * 16 + half * 8;
#pragma unroll
        for (int t = 0; t < NT; ++t) {
            const int nc = nBase + t * 16 + col;
            const float bv = bias[nc];
            const float sc = AFFINE ? scale[nc] : 1.0f;
            const float sh = AFFINE ? shift[nc] : 0.0f;
#pragma unroll
            for (int r = 0; r < 8; ++r) {
                float y = acc[i][t][r] + bv;
                if (AFFINE) y = y * sc + sh;
                if (ACT == ACT_RELU)     y = fmaxf(y, 0.0f);
                else if (ACT == ACT_SIG) y = 1.0f / (1.0f + __expf(-y));
                const size_t m = (size_t)(mBase + r);
                if (WF) outF[m * ldo  + nc] = y;
                if (WH) outH[m * ldoh + nc] = (_Float16)y;
            }
        }
    }
}

template<int NT, int MT, int ACT, bool AFFINE, bool WF, bool WH>
static void run_gemm(hipStream_t s, const _Float16* A, int lda, const _Float16* Wt,
                     const float* bias, const float* scale, const float* shift,
                     float* outF, int ldo, _Float16* outH, int ldoh,
                     int M, int N, int K)
{
    dim3 grid(M / (64 * MT), N / (16 * NT), 1);
    gemm_wmma<NT, MT, ACT, AFFINE, WF, WH><<<grid, 128, 0, s>>>(
        A, lda, Wt, bias, scale, shift, outF, ldo, outH, ldoh, K);
}

// ---------------------------------------------------------------------------
// Tiny GEMM (K<=64, small N): one thread per output element. f16 A, f32 W[K,N].
// ---------------------------------------------------------------------------
__global__ void small_mm(const _Float16* __restrict__ A, int lda,
                         const float* __restrict__ W, const float* __restrict__ bias,
                         int act, float* __restrict__ outF, int ldo,
                         int M, int N, int K)
{
    int i = blockIdx.x * 256 + threadIdx.x;
    if (i >= M * N) return;
    int m = i / N, n = i % N;
    float s = bias[n];
    const _Float16* a = A + (size_t)m * lda;
    for (int k = 0; k < K; ++k) s += (float)a[k] * W[(size_t)k * N + n];
    if (act == ACT_RELU)     s = fmaxf(s, 0.0f);
    else if (act == ACT_SIG) s = 1.0f / (1.0f + __expf(-s));
    outF[(size_t)m * ldo + n] = s;
}

// Weight convert+transpose: Wt[n*K+k] = (f16) W[k*strideK + n]
__global__ void wconv(const float* __restrict__ W, _Float16* __restrict__ Wt,
                      int K, int N, int strideK)
{
    int i = blockIdx.x * 256 + threadIdx.x;
    if (i >= N * K) return;
    int n = i / K, k = i % K;
    Wt[i] = (_Float16)W[(size_t)k * strideK + n];
}

__global__ void xprep(const float* __restrict__ x, float* __restrict__ h,
                      _Float16* __restrict__ hf, int total)
{
    int i = blockIdx.x * 256 + threadIdx.x;
    if (i >= total) return;
    float v = x[i];
    h[i] = v;
    hf[i] = (_Float16)v;
}

__global__ void bnprep(const float* __restrict__ g, const float* __restrict__ b,
                       const float* __restrict__ m, const float* __restrict__ v,
                       float* __restrict__ sc, float* __restrict__ sh, int total)
{
    int i = blockIdx.x * 256 + threadIdx.x;
    if (i >= total) return;
    float s = g[i] * rsqrtf(v[i] + 1e-5f);
    sc[i] = s;
    sh[i] = b[i] - m[i] * s;
}

// wf[b, n] = feats[b, n] * cw[b, n%32] * ci[b]   (n in [0,96))
__global__ void weighted_k(const float* __restrict__ feats, const float* __restrict__ cw,
                           const float* __restrict__ ci, _Float16* __restrict__ wf, int total)
{
    int i = blockIdx.x * 256 + threadIdx.x;
    if (i >= total) return;
    int b = i / 96, n = i % 96;
    wf[i] = (_Float16)(feats[i] * cw[b * 32 + (n & 31)] * ci[b]);
}

// h = fused + gate*h ; also refresh f16 copy
__global__ void hupdate(const float* __restrict__ fused, const float* __restrict__ gate,
                        float* __restrict__ h, _Float16* __restrict__ hf, int total)
{
    int i = blockIdx.x * 256 + threadIdx.x;
    if (i >= total) return;
    float v = fused[i] + gate[i] * h[i];
    h[i] = v;
    hf[i] = (_Float16)v;
}

// Apply staffing-cap masks from x[:, D-4], x[:, D-3]
__global__ void mask_k(const float* __restrict__ x, float* __restrict__ outN,
                       float* __restrict__ outD, int B_, int C_, int D_)
{
    int i = blockIdx.x * 256 + threadIdx.x;
    if (i >= B_ * C_) return;
    int b = i / C_, c = i % C_;
    int sn = (int)x[(size_t)b * D_ + (D_ - 4)];
    int sd = (int)x[(size_t)b * D_ + (D_ - 3)];
    if (c > sn) outN[i] = -1e9f;
    if (c > sd) outD[i] = -1e9f;
}

// ---------------------------------------------------------------------------

extern "C" void kernel_launch(void* const* d_in, const int* in_sizes, int n_in,
                              void* d_out, int out_size, void* d_ws, size_t ws_size,
                              hipStream_t stream)
{
    const int B = 32768, D = 512, H = 512, C = 128, L = 3;

    const float* x    = (const float*)d_in[0];
    const float* pbw  = (const float*)d_in[1];
    const float* pbb  = (const float*)d_in[2];
    const float* paw1 = (const float*)d_in[3];
    const float* pab1 = (const float*)d_in[4];
    const float* paw2 = (const float*)d_in[5];
    const float* pab2 = (const float*)d_in[6];
    const float* piw1 = (const float*)d_in[7];
    const float* pib1 = (const float*)d_in[8];
    const float* piw2 = (const float*)d_in[9];
    const float* pib2 = (const float*)d_in[10];
    const float* pfw  = (const float*)d_in[11];
    const float* pfb  = (const float*)d_in[12];
    const float* pgw  = (const float*)d_in[13];
    const float* pgb  = (const float*)d_in[14];
    const float* fcw  = (const float*)d_in[15];
    const float* fcb  = (const float*)d_in[16];
    const float* bng  = (const float*)d_in[17];
    const float* bnb  = (const float*)d_in[18];
    const float* bnm  = (const float*)d_in[19];
    const float* bnv  = (const float*)d_in[20];
    const float* nhw1 = (const float*)d_in[21];
    const float* nhb1 = (const float*)d_in[22];
    const float* nhw2 = (const float*)d_in[23];
    const float* nhb2 = (const float*)d_in[24];
    const float* dhw1 = (const float*)d_in[25];
    const float* dhb1 = (const float*)d_in[26];
    const float* dhw2 = (const float*)d_in[27];
    const float* dhb2 = (const float*)d_in[28];
    const float* wtw1 = (const float*)d_in[29];
    const float* wtb1 = (const float*)d_in[30];
    const float* wtw2 = (const float*)d_in[31];
    const float* wtb2 = (const float*)d_in[32];

    // ---- workspace carve-up (256B aligned) ----
    size_t off = 0;
    auto alloc = [&](size_t bytes) -> void* {
        size_t o = (off + 255) & ~(size_t)255;
        off = o + bytes;
        return (void*)((char*)d_ws + o);
    };
    float*    h     = (float*)   alloc((size_t)B * D * 4);
    _Float16* hfA   = (_Float16*)alloc((size_t)B * D * 2);
    _Float16* hfB   = (_Float16*)alloc((size_t)B * D * 2);
    _Float16* t64   = (_Float16*)alloc((size_t)B * 64 * 2);
    _Float16* t16   = (_Float16*)alloc((size_t)B * 16 * 2);
    float*    cw    = (float*)   alloc((size_t)B * 32 * 4);
    float*    ci    = (float*)   alloc((size_t)B * 4);
    float*    feats = (float*)   alloc((size_t)B * 96 * 4);
    _Float16* wf16  = (_Float16*)alloc((size_t)B * 96 * 2);
    float*    fused = (float*)   alloc((size_t)B * D * 4);
    float*    gate  = (float*)   alloc((size_t)B * D * 4);
    float*    bnsc  = (float*)   alloc((size_t)L * H * 4);
    float*    bnsh  = (float*)   alloc((size_t)L * H * 4);
    _Float16* paw1T = (_Float16*)alloc((size_t)L * 64 * D * 2);
    _Float16* piw1T = (_Float16*)alloc((size_t)L * 16 * D * 2);
    _Float16* pbwT  = (_Float16*)alloc((size_t)L * 3 * 32 * D * 2);
    _Float16* pfwT  = (_Float16*)alloc((size_t)L * D * 96 * 2);
    _Float16* pgwT  = (_Float16*)alloc((size_t)L * D * D * 2);
    _Float16* fcwT  = (_Float16*)alloc((size_t)L * H * D * 2);
    _Float16* nhw1T = (_Float16*)alloc((size_t)64 * H * 2);
    _Float16* dhw1T = (_Float16*)alloc((size_t)64 * H * 2);
    _Float16* wtw1T = (_Float16*)alloc((size_t)64 * H * 2);
    _Float16* nhw2T = (_Float16*)alloc((size_t)C * 64 * 2);
    _Float16* dhw2T = (_Float16*)alloc((size_t)C * 64 * 2);
    (void)ws_size;

    auto conv = [&](const float* W, _Float16* Wt, int K, int N, int strideK) {
        int total = N * K;
        wconv<<<(total + 255) / 256, 256, 0, stream>>>(W, Wt, K, N, strideK);
    };

    // ---- prep: input copy/convert, BN affine fold, weight f16 transposes ----
    xprep<<<(B * D + 255) / 256, 256, 0, stream>>>(x, h, hfA, B * D);
    bnprep<<<(L * H + 255) / 256, 256, 0, stream>>>(bng, bnb, bnm, bnv, bnsc, bnsh, L * H);

    for (int l = 0; l < L; ++l) {
        conv(paw1 + (size_t)l * D * 64, paw1T + (size_t)l * 64 * D, D, 64, 64);
        conv(piw1 + (size_t)l * D * 16, piw1T + (size_t)l * 16 * D, D, 16, 16);
        for (int k = 0; k < 3; ++k)
            conv(pbw + (size_t)(l * 3 + k) * D * 32, pbwT + (size_t)(l * 3 + k) * 32 * D, D, 32, 32);
        conv(pfw + (size_t)l * 96 * D, pfwT + (size_t)l * D * 96, 96, D, D);
        conv(pgw + (size_t)l * D * D, pgwT + (size_t)l * D * D, D, D, D);
        conv(fcw + (size_t)l * D * H, fcwT + (size_t)l * H * D, D, H, H);
    }
    conv(nhw1, nhw1T, H, 64, 64);
    conv(dhw1, dhw1T, H, 64, 64);
    conv(wtw1, wtw1T, H, 64, 64);
    conv(nhw2, nhw2T, 64, C, C);
    conv(dhw2, dhw2T, 64, C, C);

    // ---- ClinicalPAN stack ----
    for (int l = 0; l < L; ++l) {
        // attention branch: relu(h@paw1+b) -> sigmoid(.@paw2+b)
        run_gemm<4, 2, ACT_RELU, false, false, true>(
            stream, hfA, D, paw1T + (size_t)l * 64 * D, pab1 + l * 64,
            nullptr, nullptr, nullptr, 0, t64, 64, B, 64, D);
        small_mm<<<(B * 32 + 255) / 256, 256, 0, stream>>>(
            t64, 64, paw2 + (size_t)l * 64 * 32, pab2 + l * 32, ACT_SIG, cw, 32, B, 32, 64);
        // importance branch: relu(h@piw1+b) -> sigmoid(.@piw2+b)
        run_gemm<1, 2, ACT_RELU, false, false, true>(
            stream, hfA, D, piw1T + (size_t)l * 16 * D, pib1 + l * 16,
            nullptr, nullptr, nullptr, 0, t16, 16, B, 16, D);
        small_mm<<<(B + 255) / 256, 256, 0, stream>>>(
            t16, 16, piw2 + (size_t)l * 16, pib2 + l, ACT_SIG, ci, 1, B, 1, 16);
        // 3 feature branches -> feats[B,96]
        for (int k = 0; k < 3; ++k)
            run_gemm<2, 2, ACT_RELU, false, true, false>(
                stream, hfA, D, pbwT + (size_t)(l * 3 + k) * 32 * D,
                pbb + (size_t)(l * 3 + k) * 32, nullptr, nullptr,
                feats + k * 32, 96, nullptr, 0, B, 32, D);
        // weighted = feats * cw * ci  (f16 for next GEMM)
        weighted_k<<<(B * 96 + 255) / 256, 256, 0, stream>>>(feats, cw, ci, wf16, B * 96);
        // fused = weighted @ pfw + pfb
        run_gemm<4, 2, ACT_NONE, false, true, false>(
            stream, wf16, 96, pfwT + (size_t)l * D * 96, pfb + l * D,
            nullptr, nullptr, fused, D, nullptr, 0, B, D, 96);
        // gate = sigmoid(h @ pgw + pgb)
        run_gemm<4, 2, ACT_SIG, false, true, false>(
            stream, hfA, D, pgwT + (size_t)l * D * D, pgb + l * D,
            nullptr, nullptr, gate, D, nullptr, 0, B, D, D);
        // h = fused + gate*h
        hupdate<<<(B * D + 255) / 256, 256, 0, stream>>>(fused, gate, h, hfA, B * D);
    }

    // ---- shared FC + BN(eval) + ReLU, ping-pong f16 activations ----
    _Float16* cur = hfA;
    _Float16* nxt = hfB;
    for (int i = 0; i < L; ++i) {
        run_gemm<4, 2, ACT_RELU, true, false, true>(
            stream, cur, D, fcwT + (size_t)i * H * D, fcb + i * H,
            bnsc + i * H, bnsh + i * H, nullptr, 0, nxt, H, B, H, D);
        _Float16* tmp = cur; cur = nxt; nxt = tmp;
    }

    // ---- heads ----
    float* outN  = (float*)d_out;
    float* outDl = outN + (size_t)B * C;
    float* outW  = outDl + (size_t)B * C;

    run_gemm<4, 2, ACT_RELU, false, false, true>(
        stream, cur, H, nhw1T, nhb1, nullptr, nullptr, nullptr, 0, t64, 64, B, 64, H);
    run_gemm<4, 2, ACT_NONE, false, true, false>(
        stream, t64, 64, nhw2T, nhb2, nullptr, nullptr, outN, C, nullptr, 0, B, C, 64);

    run_gemm<4, 2, ACT_RELU, false, false, true>(
        stream, cur, H, dhw1T, dhb1, nullptr, nullptr, nullptr, 0, t64, 64, B, 64, H);
    run_gemm<4, 2, ACT_NONE, false, true, false>(
        stream, t64, 64, dhw2T, dhb2, nullptr, nullptr, outDl, C, nullptr, 0, B, C, 64);

    run_gemm<4, 2, ACT_RELU, false, false, true>(
        stream, cur, H, wtw1T, wtb1, nullptr, nullptr, nullptr, 0, t64, 64, B, 64, H);
    small_mm<<<(B + 255) / 256, 256, 0, stream>>>(
        t64, 64, wtw2, wtb2, ACT_NONE, outW, 1, B, 1, 64);

    // ---- staffing-cap masks ----
    mask_k<<<(B * C + 255) / 256, 256, 0, stream>>>(x, outN, outDl, B, C, D);
}
